// BinarizedLinear_5162550690250
// MI455X (gfx1250) — matrix operand order repaired and checked by reading, not compiled, not run on hardware
//
#include <hip/hip_runtime.h>
#include <stdint.h>

typedef __attribute__((ext_vector_type(16))) _Float16 v16h;
typedef __attribute__((ext_vector_type(4)))  _Float16 v4h;
typedef __attribute__((ext_vector_type(8)))  float    v8f;

#define K_DIM  4096
#define K32    (K_DIM / 32)     // number of BK=32 steps
#define BM     128
#define BN     128
#define BK     32
#define THREADS 256

__device__ __forceinline__ _Float16 signh(float v) {
    return (_Float16)((v > 0.f) ? 1.f : ((v < 0.f) ? -1.f : 0.f));
}

// ---------------------------------------------------------------------------
// Async copy: one 16B chunk global -> LDS via CDNA5 async-to-LDS path.
// Generic pointers to __shared__ carry the LDS byte offset in their low 32
// bits (LDS aperture truncation, ISA 10.2), so we pass that as VDST.
// ---------------------------------------------------------------------------
__device__ __forceinline__ void async_cp16(void* ldsPtr, const void* gPtr) {
    uint32_t l = (uint32_t)(uintptr_t)ldsPtr;
    asm volatile("global_load_async_to_lds_b128 %0, %1, off"
                 :: "v"(l), "v"(gPtr) : "memory");
}
__device__ __forceinline__ void wait_async0() {
    asm volatile("s_wait_asynccnt 0" ::: "memory");
}

// ---------------------------------------------------------------------------
// Pack kernels: fp32 -> f16 (x) / sign-f16 (w), written in fragment-packed,
// bank-deswizzled layout:
//   block(frag, kt) = 1024 bytes, 64 chunks of 16B.
//   element (r16, kl):  A: lane = 16*((kl>>3)&1) + r16, slot = (kl&7)+8*(kl>>4)
//                       B: lane = 16*((kl>>3)&1) + r16, slot = (kl&7)+8*(kl>>4)
//   byte = (slot>=8 ? 512 : 0) + lane*16 + (slot&7)*2
// ---------------------------------------------------------------------------
__global__ __launch_bounds__(THREADS)
void pack_a(const float* __restrict__ x, _Float16* __restrict__ apk) {
    const int f   = blockIdx.x * THREADS + threadIdx.x;   // float4 index
    const int row = f >> 10;                              // K/4 = 1024 per row
    const int k0  = (f & 1023) << 2;
    float4 v = *reinterpret_cast<const float4*>(x + (size_t)row * K_DIM + k0);
    v4h h;
    h[0] = (_Float16)v.x; h[1] = (_Float16)v.y;
    h[2] = (_Float16)v.z; h[3] = (_Float16)v.w;
    const int mfrag = row >> 4, m16 = row & 15;
    const int kt = k0 >> 5,  kl = k0 & 31;
    const int lane = (((kl >> 3) & 1) << 4) + m16;
    const int slot = (kl & 7) + ((kl >> 4) << 3);
    const size_t byteOff = ((size_t)(mfrag * K32 + kt)) * 1024
                         + (size_t)((slot >= 8) ? 512 : 0) + lane * 16 + (slot & 7) * 2;
    *reinterpret_cast<v4h*>((char*)apk + byteOff) = h;
}

__global__ __launch_bounds__(THREADS)
void pack_b(const float* __restrict__ w, _Float16* __restrict__ bpk) {
    const int f   = blockIdx.x * THREADS + threadIdx.x;
    const int row = f >> 10;                              // n index
    const int k0  = (f & 1023) << 2;
    float4 v = *reinterpret_cast<const float4*>(w + (size_t)row * K_DIM + k0);
    v4h h;
    h[0] = signh(v.x); h[1] = signh(v.y);
    h[2] = signh(v.z); h[3] = signh(v.w);
    const int nfrag = row >> 4, n16 = row & 15;
    const int kt = k0 >> 5,  kl = k0 & 31;
    const int lane = (((kl >> 3) & 1) << 4) + n16;
    const int slot = (kl & 7) + ((kl >> 4) << 3);
    const size_t byteOff = ((size_t)(nfrag * K32 + kt)) * 1024
                         + (size_t)((slot >= 8) ? 512 : 0) + lane * 16 + (slot & 7) * 2;
    *reinterpret_cast<v4h*>((char*)bpk + byteOff) = h;
}

// Reassemble a v16h fragment from its two 16B chunks (lo: slots 0-7, hi: 8-15)
__device__ __forceinline__ v16h load_frag(const _Float16* fragBase, int lane) {
    union { uint4 q[2]; v16h h; } t;
    const char* p = reinterpret_cast<const char*>(fragBase);
    t.q[0] = *reinterpret_cast<const uint4*>(p + lane * 16);
    t.q[1] = *reinterpret_cast<const uint4*>(p + 512 + lane * 16);
    return t.h;
}

// ---------------------------------------------------------------------------
// Main GEMM on packed f16 operands: async global->LDS staging + WMMA
// ---------------------------------------------------------------------------
__global__ __launch_bounds__(THREADS)
void gemm_packed(const _Float16* __restrict__ apk,
                 const _Float16* __restrict__ bpk,
                 float* __restrict__ out, int N) {
    __shared__ _Float16 ldsA[2][BM * BK];   // 8 frag-blocks of 1KB each
    __shared__ _Float16 ldsB[2][BK * BN];

    const int tid    = threadIdx.x;
    const int lane   = tid & 31;
    const int wave   = tid >> 5;
    const int wave_m = wave & 3;    // M frags 2*wave_m .. +1
    const int wave_n = wave >> 2;   // N frags 4*wave_n .. +3

    const int mfrag0 = blockIdx.y * 8;
    const int nfrag0 = blockIdx.x * 8;

    // stage one BK tile: 8 A-frag blocks + 8 B-frag blocks = 1024 chunks of 16B
    auto copy_tile = [&](int kt, int buf) {
#pragma unroll
        for (int i = 0; i < 4; ++i) {
            const int  c    = tid + THREADS * i;      // 0..1023
            const bool isA  = (i < 2);                // first 512 chunks = A
            const int  cc   = isA ? c : (c - 512);
            const int  frag = cc >> 6;
            const int  off  = (cc & 63) << 4;
            const size_t gbyte =
                ((size_t)((isA ? mfrag0 : nfrag0) + frag) * K32 + kt) * 1024 + off;
            const char* g = (const char*)(isA ? (const void*)apk : (const void*)bpk) + gbyte;
            char* l = (char*)(isA ? (void*)ldsA[buf] : (void*)ldsB[buf]) + frag * 1024 + off;
            async_cp16(l, g);
        }
    };

    v8f acc[2][4];
#pragma unroll
    for (int mi = 0; mi < 2; ++mi)
#pragma unroll
        for (int ni = 0; ni < 4; ++ni)
            acc[mi][ni] = (v8f){0.f, 0.f, 0.f, 0.f, 0.f, 0.f, 0.f, 0.f};

    // prologue: tile 0
    copy_tile(0, 0);
    wait_async0();
    __syncthreads();

    for (int it = 0; it < K32; ++it) {
        const int cur = it & 1;

        // stage next tile directly into the other LDS buffer (overlaps WMMA)
        if (it + 1 < K32)
            copy_tile(it + 1, cur ^ 1);

        const _Float16* lA = ldsA[cur];
        const _Float16* lB = ldsB[cur];
        v16h af[2], bf[4];
#pragma unroll
        for (int mi = 0; mi < 2; ++mi)
            af[mi] = load_frag(lA + (2 * wave_m + mi) * 512, lane);
#pragma unroll
        for (int ni = 0; ni < 4; ++ni)
            bf[ni] = load_frag(lB + (4 * wave_n + ni) * 512, lane);

#pragma unroll
        for (int mi = 0; mi < 2; ++mi)
#pragma unroll
            for (int ni = 0; ni < 4; ++ni)
                acc[mi][ni] = __builtin_amdgcn_wmma_f32_16x16x32_f16(
                    false, af[mi], false, bf[ni],
                    (short)0, acc[mi][ni], false, false);

        if (it + 1 < K32) {
            wait_async0();       // own async writes to nxt landed
            __syncthreads();     // everyone's landed; also fences reads of cur
        }
    }

    // epilogue: D layout (VGPR j = row m%8; lanes: N=lane&15, M-half=lane>>4)
    const int bm0 = blockIdx.y * BM;
    const int bn0 = blockIdx.x * BN;
    const int r8  = (lane >> 4) << 3;
    const int cn  = lane & 15;
#pragma unroll
    for (int mi = 0; mi < 2; ++mi) {
        const int gm = bm0 + (2 * wave_m + mi) * 16 + r8;
#pragma unroll
        for (int ni = 0; ni < 4; ++ni) {
            const int gn = bn0 + (4 * wave_n + ni) * 16 + cn;
#pragma unroll
            for (int j = 0; j < 8; ++j)
                out[(size_t)(gm + j) * N + gn] = acc[mi][ni][j];
        }
    }
}

// ---------------------------------------------------------------------------
// Fallback: fused single-kernel version (round-1) if workspace is too small.
// ---------------------------------------------------------------------------
#define STORE_TILE(BUF)                                                        \
    do {                                                                       \
        _Pragma("unroll")                                                      \
        for (int i = 0; i < 4; ++i) {                                          \
            const int row = rowi[i];                                           \
            const int k0  = c4i[i] << 2;                                       \
            const int frg = row >> 4;                                          \
            const int m   = row & 15;                                          \
            const int laneA = (((k0 >> 3) & 1) << 4) + m;                      \
            const int slotA = (k0 & 7) + ((k0 >> 4) << 3);                     \
            v4h ha;                                                            \
            ha[0] = (_Float16)ax[i].x; ha[1] = (_Float16)ax[i].y;              \
            ha[2] = (_Float16)ax[i].z; ha[3] = (_Float16)ax[i].w;              \
            *reinterpret_cast<v4h*>(&fldsA[BUF][frg * 512 + laneA * 16 + slotA]) = ha; \
            const int laneB = (((k0 >> 3) & 1) << 4) + m;                      \
            const int slotB = (k0 & 7) + ((k0 >> 4) << 3);                     \
            v4h hb;                                                            \
            hb[0] = signh(bx[i].x); hb[1] = signh(bx[i].y);                    \
            hb[2] = signh(bx[i].z); hb[3] = signh(bx[i].w);                    \
            *reinterpret_cast<v4h*>(&fldsB[BUF][frg * 512 + laneB * 16 + slotB]) = hb; \
        }                                                                      \
    } while (0)

__global__ __launch_bounds__(THREADS)
void binlin_wmma_fused(const float* __restrict__ x, const float* __restrict__ w,
                       float* __restrict__ out, int M, int N, int K) {
    __shared__ _Float16 fldsA[2][BM * BK];
    __shared__ _Float16 fldsB[2][BK * BN];

    const int tid = threadIdx.x, lane = tid & 31, wave = tid >> 5;
    const int wave_m = wave & 3, wave_n = wave >> 2;
    const int bm0 = blockIdx.y * BM, bn0 = blockIdx.x * BN;

    int rowi[4], c4i[4];
#pragma unroll
    for (int i = 0; i < 4; ++i) {
        const int f = tid + THREADS * i;
        rowi[i] = f >> 3; c4i[i] = f & 7;
    }

    v8f acc[2][4];
#pragma unroll
    for (int mi = 0; mi < 2; ++mi)
#pragma unroll
        for (int ni = 0; ni < 4; ++ni)
            acc[mi][ni] = (v8f){0.f,0.f,0.f,0.f,0.f,0.f,0.f,0.f};

    float4 ax[4], bx[4];
#pragma unroll
    for (int i = 0; i < 4; ++i) {
        ax[i] = *reinterpret_cast<const float4*>(x + (size_t)(bm0 + rowi[i]) * K + (c4i[i] << 2));
        bx[i] = *reinterpret_cast<const float4*>(w + (size_t)(bn0 + rowi[i]) * K + (c4i[i] << 2));
    }
    STORE_TILE(0);
    __syncthreads();

    const int NT = K / BK;
    for (int it = 0; it < NT; ++it) {
        const int cur = it & 1;
        if (it + 1 < NT) {
            const int kt = (it + 1) * BK;
#pragma unroll
            for (int i = 0; i < 4; ++i) {
                ax[i] = *reinterpret_cast<const float4*>(x + (size_t)(bm0 + rowi[i]) * K + kt + (c4i[i] << 2));
                bx[i] = *reinterpret_cast<const float4*>(w + (size_t)(bn0 + rowi[i]) * K + kt + (c4i[i] << 2));
            }
        }
        v16h af[2], bf[4];
#pragma unroll
        for (int mi = 0; mi < 2; ++mi)
            af[mi] = *reinterpret_cast<const v16h*>(&fldsA[cur][(2 * wave_m + mi) * 512 + lane * 16]);
#pragma unroll
        for (int ni = 0; ni < 4; ++ni)
            bf[ni] = *reinterpret_cast<const v16h*>(&fldsB[cur][(4 * wave_n + ni) * 512 + lane * 16]);
#pragma unroll
        for (int mi = 0; mi < 2; ++mi)
#pragma unroll
            for (int ni = 0; ni < 4; ++ni)
                acc[mi][ni] = __builtin_amdgcn_wmma_f32_16x16x32_f16(
                    false, af[mi], false, bf[ni], (short)0, acc[mi][ni], false, false);
        if (it + 1 < NT) {
            __syncthreads();
            STORE_TILE(cur ^ 1);
            __syncthreads();
        }
    }

    const int r8 = (lane >> 4) << 3, cn = lane & 15;
#pragma unroll
    for (int mi = 0; mi < 2; ++mi) {
        const int gm = bm0 + (2 * wave_m + mi) * 16 + r8;
#pragma unroll
        for (int ni = 0; ni < 4; ++ni) {
            const int gn = bn0 + (4 * wave_n + ni) * 16 + cn;
#pragma unroll
            for (int j = 0; j < 8; ++j)
                out[(size_t)(gm + j) * N + gn] = acc[mi][ni][j];
        }
    }
}

extern "C" void kernel_launch(void* const* d_in, const int* in_sizes, int n_in,
                              void* d_out, int out_size, void* d_ws, size_t ws_size,
                              hipStream_t stream) {
    const float* x = (const float*)d_in[0];   // (M, K) fp32
    const float* w = (const float*)d_in[1];   // (N, K) fp32
    float* out = (float*)d_out;               // (M, N) fp32

    const int K = K_DIM;
    const int M = in_sizes[0] / K;            // 8192
    const int N = in_sizes[1] / K;            // 4096

    const size_t needA = (size_t)M * K * sizeof(_Float16);
    const size_t needB = (size_t)N * K * sizeof(_Float16);

    if (ws_size >= needA + needB) {
        _Float16* apk = (_Float16*)d_ws;
        _Float16* bpk = apk + (size_t)M * K;
        pack_a<<<(M / 1) * (K / 4) / THREADS, THREADS, 0, stream>>>(x, apk);
        pack_b<<<(N / 1) * (K / 4) / THREADS, THREADS, 0, stream>>>(w, bpk);
        dim3 grid(N / BN, M / BM);
        gemm_packed<<<grid, THREADS, 0, stream>>>(apk, bpk, out, N);
    } else {
        dim3 grid(N / BN, M / BM);
        binlin_wmma_fused<<<grid, THREADS, 0, stream>>>(x, w, out, M, N, K);
    }
}
